// Attention_62191126446357
// MI455X (gfx1250) — matrix operand order repaired and checked by reading, not compiled, not run on hardware
//
#include <hip/hip_runtime.h>

// ---------------------------------------------------------------------------
// Problem constants (match reference):  B=8, S=16, H=32, D=64, HID=2048,
// PAST=4096, T = PAST+S = 4112, M = B*S = 128 GEMM rows.
// ---------------------------------------------------------------------------
#define B_    8
#define S_    16
#define H_    32
#define D_    64
#define HID_  2048
#define PAST_ 4096
#define TTOT_ (PAST_ + S_)   // 4112
#define MROWS_ (B_ * S_)     // 128
#define NSPLIT 4

typedef __attribute__((ext_vector_type(16))) _Float16 v16h;
typedef __attribute__((ext_vector_type(8)))  _Float16 v8h;
typedef __attribute__((ext_vector_type(8)))  float    v8f;
typedef __attribute__((ext_vector_type(4)))  float    v4f;

__device__ __forceinline__ v8f wmma_f16(v16h a, v16h b, v8f c) {
  // v_wmma_f32_16x16x32_f16 : D(16x16 f32) = A(16x32 f16) * B(32x16 f16) + C
  return __builtin_amdgcn_wmma_f32_16x16x32_f16(false, a, false, b,
                                                (short)0, c, false, false);
}

// ---------------------------------------------------------------------------
// GEMM:  Y[M=128, N=2048] = X[128, 2048] @ W[2048, 2048]^T   (torch Linear)
// One wave owns a 16(M) x 64(N) tile: A fragment reused across 4 WMMAs.
// Fragment layouts used here (contiguous K per lane half, same order in A and B):
//   A 16x32 f16: lanes 0-15 row M=lane, K=kc..kc+15;
//                lanes 16-31 same rows, K=kc+16..kc+31.
//   B 32x16 f16: lanes 0-15 col N=lane, K=kc..kc+15 (VGPRj = K 2j,2j+1);
//                lanes 16-31 K=kc+16..kc+31.
// ---------------------------------------------------------------------------
__global__ __launch_bounds__(256)
void gemm_xwt(const float* __restrict__ X, const float* __restrict__ W,
              float* __restrict__ Y) {
  const int wave = blockIdx.x * (blockDim.x >> 5) + (threadIdx.x >> 5);
  const int lane = threadIdx.x & 31;
  const int lr   = lane & 15;
  const int hi   = lane >> 4;          // which 16-lane half
  const int m0   = (wave & 7) * 16;    // 8 M tiles
  const int n0   = (wave >> 3) * 64;   // 32 N blocks of 64

  v8f acc[4] = {};

  const float* xrow = X + (size_t)(m0 + lr) * HID_;
  for (int kc = 0; kc < HID_; kc += 32) {
    // ---- A fragment (shared by the 4 N tiles) ----
    const int ka = kc + hi * 16;
    v4f x0 = *(const v4f*)(xrow + ka);
    v4f x1 = *(const v4f*)(xrow + ka + 4);
    v4f x2 = *(const v4f*)(xrow + ka + 8);
    v4f x3 = *(const v4f*)(xrow + ka + 12);
    v16h a;
#pragma unroll
    for (int i = 0; i < 4; ++i) {
      a[i]      = (_Float16)x0[i];
      a[i + 4]  = (_Float16)x1[i];
      a[i + 8]  = (_Float16)x2[i];
      a[i + 12] = (_Float16)x3[i];
    }
    // ---- 4 B fragments + WMMA ----
    const int kb = kc + hi * 16;
#pragma unroll
    for (int nt = 0; nt < 4; ++nt) {
      const float* wr = W + (size_t)(n0 + nt * 16 + lr) * HID_ + kb;
      v4f w0 = *(const v4f*)(wr);
      v4f w1 = *(const v4f*)(wr + 4);
      v4f w2 = *(const v4f*)(wr + 8);
      v4f w3 = *(const v4f*)(wr + 12);
      v16h bb;
#pragma unroll
      for (int i = 0; i < 4; ++i) {
        bb[i]      = (_Float16)w0[i];
        bb[i + 4]  = (_Float16)w1[i];
        bb[i + 8]  = (_Float16)w2[i];
        bb[i + 12] = (_Float16)w3[i];
      }
      acc[nt] = wmma_f16(a, bb, acc[nt]);
    }
  }
  // C/D layout: VGPR r, lane -> row m0 + r + 8*hi, col n0 + 16*nt + lr
#pragma unroll
  for (int nt = 0; nt < 4; ++nt)
#pragma unroll
    for (int r = 0; r < 8; ++r)
      Y[(size_t)(m0 + r + 8 * hi) * HID_ + (n0 + nt * 16 + lr)] = acc[nt][r];
}

// ---------------------------------------------------------------------------
// Flash-decoding attention. Block = one (b,h); 4 waves = 4 T-splits.
// Q,KN,VN are row-major [B*S, HID] (col = h*64 + d); past K/V are [B,H,PAST,D].
// Each wave streams 32-key chunks: QK^T (2 tiles x 2 K-steps of 32 over D),
// online softmax, P restaged via LDS (C-layout -> A-layout), P@V (4 WMMAs).
// Splits merged with log-sum-exp through LDS.
// ---------------------------------------------------------------------------
__global__ __launch_bounds__(128)
void attn_flash(const float* __restrict__ Q,  const float* __restrict__ KN,
                const float* __restrict__ VN, const float* __restrict__ PK,
                const float* __restrict__ PV, float* __restrict__ O) {
  __shared__ _Float16 shP[NSPLIT][16][32];
  __shared__ float    shO[NSPLIT][16][64];
  __shared__ float    shM[NSPLIT][16];
  __shared__ float    shL[NSPLIT][16];

  const int bh   = blockIdx.x;
  const int b    = bh >> 5;          // / H_
  const int h    = bh & (H_ - 1);
  const int wave = threadIdx.x >> 5; // split id
  const int lane = threadIdx.x & 31;
  const int lr   = lane & 15;
  const int hi   = lane >> 4;

  // ---- Q A-fragments for D chunks [0,32) and [32,64), kept in registers ----
  const float* qrow = Q + (size_t)(b * S_ + lr) * HID_ + h * D_;
  v16h qa[2];
#pragma unroll
  for (int dc = 0; dc < 2; ++dc) {
    const int ko = dc * 32 + hi * 16;
    v4f x0 = *(const v4f*)(qrow + ko);
    v4f x1 = *(const v4f*)(qrow + ko + 4);
    v4f x2 = *(const v4f*)(qrow + ko + 8);
    v4f x3 = *(const v4f*)(qrow + ko + 12);
#pragma unroll
    for (int i = 0; i < 4; ++i) {
      qa[dc][i]      = (_Float16)x0[i];
      qa[dc][i + 4]  = (_Float16)x1[i];
      qa[dc][i + 8]  = (_Float16)x2[i];
      qa[dc][i + 12] = (_Float16)x3[i];
    }
  }

  // Running softmax state: m_r/l_r[r] belong to row (r + 8*hi) — exactly the
  // C/D fragment row layout, duplicated across the 16 lanes of each half.
  float m_r[8], l_r[8];
  v8f   acc[4] = {};
#pragma unroll
  for (int r = 0; r < 8; ++r) { m_r[r] = -3.0e38f; l_r[r] = 0.0f; }

  const int nchunks = (TTOT_ + 31) / 32;  // 129 (last chunk: 16 valid keys)
  for (int c = wave; c < nchunks; c += NSPLIT) {
    const int t0 = c * 32;
    const float *kb_, *vb_;
    size_t kst;
    if (t0 < PAST_) {  // chunk boundary 4096 is 32-aligned -> uniform select
      const size_t base = ((size_t)(b * H_ + h) * PAST_ + t0) * D_;
      kb_ = PK + base; vb_ = PV + base; kst = D_;
    } else {
      const size_t base = (size_t)(b * S_ + (t0 - PAST_)) * HID_ + h * D_;
      kb_ = KN + base; vb_ = VN + base; kst = HID_;
    }

    // Prefetch this wave's next chunk of the KV stream (global_prefetch_b8)
    if (t0 + NSPLIT * 32 + 32 <= PAST_) {
      const size_t nb = ((size_t)(b * H_ + h) * PAST_ + t0 + NSPLIT * 32 + lane) * D_;
      __builtin_prefetch(PK + nb, 0, 1);
      __builtin_prefetch(PV + nb, 0, 1);
    }

    // ---- scores: two 16x16 tiles, K-dim = D = 64 (2 WMMA steps) ----
    v8f   p[2];
    float cmax[8];
#pragma unroll
    for (int r = 0; r < 8; ++r) cmax[r] = -3.0e38f;
#pragma unroll
    for (int nt = 0; nt < 2; ++nt) {
      const int tcol = t0 + nt * 16 + lr;   // key position for this lane
      v16h kf0 = {}, kf1 = {};
      if (tcol < TTOT_) {                   // guard loads only; WMMA below
        const float* kr = kb_ + (size_t)(nt * 16 + lr) * kst;
        {
          const int kofs = hi * 16;         // d 0..31
          v4f w0 = *(const v4f*)(kr + kofs);
          v4f w1 = *(const v4f*)(kr + kofs + 4);
          v4f w2 = *(const v4f*)(kr + kofs + 8);
          v4f w3 = *(const v4f*)(kr + kofs + 12);
#pragma unroll
          for (int i = 0; i < 4; ++i) {
            kf0[i] = (_Float16)w0[i];     kf0[i + 4]  = (_Float16)w1[i];
            kf0[i + 8] = (_Float16)w2[i]; kf0[i + 12] = (_Float16)w3[i];
          }
        }
        {
          const int kofs = 32 + hi * 16;    // d 32..63
          v4f w0 = *(const v4f*)(kr + kofs);
          v4f w1 = *(const v4f*)(kr + kofs + 4);
          v4f w2 = *(const v4f*)(kr + kofs + 8);
          v4f w3 = *(const v4f*)(kr + kofs + 12);
#pragma unroll
          for (int i = 0; i < 4; ++i) {
            kf1[i] = (_Float16)w0[i];     kf1[i + 4]  = (_Float16)w1[i];
            kf1[i + 8] = (_Float16)w2[i]; kf1[i + 12] = (_Float16)w3[i];
          }
        }
      }
      v8f sc = {};
      sc = wmma_f16(qa[0], kf0, sc);   // EXEC all-ones here (reconverged)
      sc = wmma_f16(qa[1], kf1, sc);
      const float mask = (tcol < TTOT_) ? 0.0f : -3.0e38f;  // -inf padding
#pragma unroll
      for (int r = 0; r < 8; ++r) {
        const float v = sc[r] * 0.125f + mask;  // 1/sqrt(64)
        p[nt][r] = v;
        cmax[r]  = fmaxf(cmax[r], v);
      }
    }
    // Row max across the 16-lane half (xor masks <=8 stay within each half)
#pragma unroll
    for (int off = 8; off >= 1; off >>= 1)
#pragma unroll
      for (int r = 0; r < 8; ++r)
        cmax[r] = fmaxf(cmax[r], __shfl_xor(cmax[r], off, 32));

    float sf[8], rs[8];
#pragma unroll
    for (int r = 0; r < 8; ++r) {
      const float mn = fmaxf(m_r[r], cmax[r]);
      sf[r]   = __expf(m_r[r] - mn);
      m_r[r]  = mn;
      p[0][r] = __expf(p[0][r] - mn);
      p[1][r] = __expf(p[1][r] - mn);
      rs[r]   = p[0][r] + p[1][r];
    }
#pragma unroll
    for (int off = 8; off >= 1; off >>= 1)
#pragma unroll
      for (int r = 0; r < 8; ++r)
        rs[r] += __shfl_xor(rs[r], off, 32);
#pragma unroll
    for (int r = 0; r < 8; ++r) l_r[r] = l_r[r] * sf[r] + rs[r];
#pragma unroll
    for (int c4 = 0; c4 < 4; ++c4)
#pragma unroll
      for (int r = 0; r < 8; ++r) acc[c4][r] *= sf[r];

    // ---- P: C-layout -> LDS -> A-layout (same-wave RAW, DScnt handled) ----
#pragma unroll
    for (int nt = 0; nt < 2; ++nt)
#pragma unroll
      for (int r = 0; r < 8; ++r)
        shP[wave][r + 8 * hi][nt * 16 + lr] = (_Float16)p[nt][r];
    v16h pa;
    {
      const int co = hi * 16;
      v8h a0 = *(const v8h*)(&shP[wave][lr][co]);
      v8h a1 = *(const v8h*)(&shP[wave][lr][co + 8]);
#pragma unroll
      for (int i = 0; i < 8; ++i) { pa[i] = a0[i]; pa[i + 8] = a1[i]; }
    }

    // ---- O += P @ V : 4 N tiles of 16 over D=64 ----
    const bool   okv  = (t0 + hi * 16) < TTOT_;
    const float* vrow = vb_ + (size_t)(hi * 16) * kst;
#pragma unroll
    for (int c4 = 0; c4 < 4; ++c4) {
      v16h vf = {};
      if (okv) {
#pragma unroll
        for (int k = 0; k < 16; ++k)   // coalesced across lanes (64B rows)
          vf[k] = (_Float16)vrow[(size_t)k * kst + c4 * 16 + lr];
      }
      acc[c4] = wmma_f16(pa, vf, acc[c4]);
    }
  }

  // ---- publish per-split partials ----
#pragma unroll
  for (int c4 = 0; c4 < 4; ++c4)
#pragma unroll
    for (int r = 0; r < 8; ++r)
      shO[wave][r + 8 * hi][c4 * 16 + lr] = acc[c4][r];
  if (lr == 0) {
#pragma unroll
    for (int r = 0; r < 8; ++r) {
      shM[wave][r + 8 * hi] = m_r[r];
      shL[wave][r + 8 * hi] = l_r[r];
    }
  }
  __syncthreads();

  // ---- log-sum-exp merge of the 4 splits: 128 threads x 8 elems = 16x64 ----
#pragma unroll
  for (int j = 0; j < 8; ++j) {
    const int e = threadIdx.x * 8 + j;
    const int s = e >> 6;
    const int d = e & 63;
    float mm = -3.0e38f;
#pragma unroll
    for (int sp = 0; sp < NSPLIT; ++sp) mm = fmaxf(mm, shM[sp][s]);
    float den = 0.0f, num = 0.0f;
#pragma unroll
    for (int sp = 0; sp < NSPLIT; ++sp) {
      const float w = __expf(shM[sp][s] - mm);
      den += shL[sp][s] * w;
      num += shO[sp][s][d] * w;
    }
    O[(size_t)(b * S_ + s) * HID_ + h * D_ + d] = num / den;
  }
}

// ---------------------------------------------------------------------------
// Launch: 3 QKV GEMMs -> flash attention (1024 waves) -> output GEMM.
// Workspace: q, k_new, v_new, attn  (4 x 128*2048 f32 = 4 MB).
// ---------------------------------------------------------------------------
extern "C" void kernel_launch(void* const* d_in, const int* in_sizes, int n_in,
                              void* d_out, int out_size, void* d_ws, size_t ws_size,
                              hipStream_t stream) {
  (void)in_sizes; (void)n_in; (void)out_size; (void)ws_size;
  const float* hs = (const float*)d_in[0];
  const float* pk = (const float*)d_in[1];
  const float* pv = (const float*)d_in[2];
  const float* wq = (const float*)d_in[3];
  const float* wk = (const float*)d_in[4];
  const float* wv = (const float*)d_in[5];
  const float* wo = (const float*)d_in[6];

  float* q  = (float*)d_ws;
  float* kn = q  + (size_t)MROWS_ * HID_;
  float* vn = kn + (size_t)MROWS_ * HID_;
  float* at = vn + (size_t)MROWS_ * HID_;

  // 256 waves per GEMM: 32 blocks x 8 waves (one N-block of 64 per block)
  gemm_xwt<<<32, 256, 0, stream>>>(hs, wq, q);
  gemm_xwt<<<32, 256, 0, stream>>>(hs, wk, kn);
  gemm_xwt<<<32, 256, 0, stream>>>(hs, wv, vn);
  attn_flash<<<B_ * H_, 128, 0, stream>>>(q, kn, vn, pk, pv, at);
  gemm_xwt<<<32, 256, 0, stream>>>(at, wo, (float*)d_out);
}
